// BiMambaLayer_21775484190905
// MI455X (gfx1250) — compile-verified, matrix operand order checked by reading, not run on hardware
//
#include <hip/hip_runtime.h>
#include <hip/hip_bf16.h>
#include <cstdint>

#define B_SZ   4
#define SEQ_L  2048
#define DM     512
#define DI     1024
#define DSTATE 16
#define RK     32
#define MTOT   (B_SZ * SEQ_L)   /* 8192 tokens */
#define NCHUNK 32
#define CLEN   (SEQ_L / NCHUNK) /* 64 */

typedef _Float16 h16 __attribute__((ext_vector_type(16)));
typedef _Float16 h8  __attribute__((ext_vector_type(8)));
typedef float    f8  __attribute__((ext_vector_type(8)));
typedef int      iv4 __attribute__((ext_vector_type(4)));

#define AS1 __attribute__((address_space(1)))
#define AS3 __attribute__((address_space(3)))

#if defined(__has_builtin)
#if __has_builtin(__builtin_amdgcn_global_load_async_to_lds_b128)
#define HAVE_ASYNC_LDS 1
#endif
#endif
#ifndef HAVE_ASYNC_LDS
#define HAVE_ASYNC_LDS 0
#endif

// global -> LDS 16-byte copy: async DMA path (ASYNCcnt) when available.
__device__ __forceinline__ void cp_g2l_b128(const _Float16* g, _Float16* l) {
#if HAVE_ASYNC_LDS
  __builtin_amdgcn_global_load_async_to_lds_b128((AS1 iv4*)g, (AS3 iv4*)l, 0, 0);
#else
  *(h8*)l = *(const h8*)g;
#endif
}

// Allow the newest N async-load instructions to stay in flight (in-order
// completion => everything older is done).
template<int N>
__device__ __forceinline__ void wait_async_lds_n() {
#if HAVE_ASYNC_LDS
#if __has_builtin(__builtin_amdgcn_s_wait_asynccnt)
  __builtin_amdgcn_s_wait_asynccnt(N);
#else
  asm volatile("s_wait_asynccnt %0" :: "i"(N) : "memory");
#endif
#endif
}

// ---------------------------------------------------------------------------
// Fragment load: CDNA5 16-bit A/B layout = lane holds row (l&15), two
// contiguous 8-half runs at k = (l>>4)*8 and k+16 (CDNA5 ISA).
// ---------------------------------------------------------------------------
__device__ __forceinline__ h16 frag_ld(const _Float16* p) {
  h8 lo = *(const h8*)p;
  h8 hi = *(const h8*)(p + 16);
  return __builtin_shufflevector(lo, hi, 0,1,2,3,4,5,6,7,8,9,10,11,12,13,14,15);
}

// ---------------------------------------------------------------------------
// WMMA GEMM:  C[M,N] = A[M,K](f16) * W[N,K]^T(f16), f32 accumulate.
// Block: 128 x BN C tile, 256 threads = 8 wave32 (4 over M, 2 over N).
// Each wave: 32 x (BN/2)  ->  2 x (BN/32) accumulators of 16x16.
// K staged in TRIPLE-buffered LDS 32-deep slices via async global->LDS;
// s_wait_asynccnt <= 2*CHUNKS keeps two slices of loads in flight.
// ---------------------------------------------------------------------------
template<int BN, bool HALF_OUT>
__global__ __launch_bounds__(256) void gemm_wmma(
    const _Float16* __restrict__ A, const _Float16* __restrict__ W,
    void* __restrict__ Cout, int K, int lda, int ldw, int ldc)
{
  constexpr int NT     = BN / 32;       // n-tiles per wave (4 or 2)
  constexpr int CHUNKS = 2 + BN / 64;   // async instrs per slice per thread
  __shared__ alignas(16) _Float16 lA[3][128 * 32];
  __shared__ alignas(16) _Float16 lB[3][BN * 32];

  const int tid  = threadIdx.x;
  const int lane = tid & 31;
  const int wid  = tid >> 5;
  const int wm   = wid & 3;             // 4 waves over M
  const int wn   = wid >> 2;            // 2 waves over N
  const int bm   = blockIdx.x * 128;
  const int bn   = blockIdx.y * BN;

  const int r  = lane & 15;
  const int kb = (lane >> 4) * 8;

  f8 acc[2][NT] = {};

  auto load_slice = [&](int k0, int buf) {
    #pragma unroll
    for (int it = 0; it < 2; ++it) {    // A: 512 chunks -> 2/thread
      int i = tid + it * 256;
      int row = i >> 2, ch = i & 3;
      cp_g2l_b128(&A[(size_t)(bm + row) * lda + k0 + ch * 8],
                  &lA[buf][row * 32 + ch * 8]);
    }
    #pragma unroll
    for (int it = 0; it < BN / 64; ++it) {  // W: BN*4 chunks
      int i = tid + it * 256;
      int row = i >> 2, ch = i & 3;
      cp_g2l_b128(&W[(size_t)(bn + row) * ldw + k0 + ch * 8],
                  &lB[buf][row * 32 + ch * 8]);
    }
  };

  const int nslices = K >> 5;
  load_slice(0, 0);
  if (nslices > 1) load_slice(32, 1);
  wait_async_lds_n<CHUNKS>();           // slice 0 resident, slice 1 may fly
  __syncthreads();

  for (int s = 0; s < nslices; ++s) {
    const int cur = s % 3;
    if (s + 2 < nslices) load_slice((s + 2) * 32, (s + 2) % 3);

    h16 af[2], bf[NT];
    #pragma unroll
    for (int it = 0; it < 2; ++it)
      af[it] = frag_ld(&lA[cur][(wm * 32 + it * 16 + r) * 32 + kb]);
    #pragma unroll
    for (int jt = 0; jt < NT; ++jt)
      bf[jt] = frag_ld(&lB[cur][(wn * (BN / 2) + jt * 16 + r) * 32 + kb]);

    #pragma unroll
    for (int it = 0; it < 2; ++it)
      #pragma unroll
      for (int jt = 0; jt < NT; ++jt)
        acc[it][jt] = __builtin_amdgcn_wmma_f32_16x16x32_f16(
            false, af[it], false, bf[jt], (short)0, acc[it][jt], false, false);

    wait_async_lds_n<2 * CHUNKS>();     // slices s+1 and s+2 may still be in flight
    __syncthreads();
  }

  // C/D layout: VGPR rr holds (M = rr + 8*(lane>>4), N = lane&15) of the tile.
  #pragma unroll
  for (int it = 0; it < 2; ++it)
    #pragma unroll
    for (int jt = 0; jt < NT; ++jt) {
      const int row0 = bm + wm * 32 + it * 16 + (lane >> 4) * 8;
      const int col  = bn + wn * (BN / 2) + jt * 16 + (lane & 15);
      f8 v = acc[it][jt];
      #pragma unroll
      for (int rr = 0; rr < 8; ++rr) {
        size_t off = (size_t)(row0 + rr) * ldc + col;
        if constexpr (HALF_OUT) ((_Float16*)Cout)[off] = (_Float16)v[rr];
        else                    ((float*)Cout)[off]    = v[rr];
      }
    }
}

// ---------------------------------------------------------------------------
// LN over D_MODEL=512, one token per block; emits f16 plus L-flipped copy.
// ---------------------------------------------------------------------------
__global__ __launch_bounds__(256) void ln_in_kernel(
    const float* __restrict__ x, const float* __restrict__ w, const float* __restrict__ b,
    _Float16* __restrict__ xn, _Float16* __restrict__ xnrev)
{
  const int token = blockIdx.x;
  const int bb = token / SEQ_L, l = token % SEQ_L;
  const int tid = threadIdx.x;
  const float* xr = x + (size_t)token * DM;
  float v0 = xr[tid], v1 = xr[tid + 256];
  __shared__ float red[256];
  red[tid] = v0 + v1; __syncthreads();
  for (int s = 128; s > 0; s >>= 1) { if (tid < s) red[tid] += red[tid + s]; __syncthreads(); }
  const float mean = red[0] * (1.f / DM);
  __syncthreads();
  float d0 = v0 - mean, d1 = v1 - mean;
  red[tid] = d0 * d0 + d1 * d1; __syncthreads();
  for (int s = 128; s > 0; s >>= 1) { if (tid < s) red[tid] += red[tid + s]; __syncthreads(); }
  const float rstd = rsqrtf(red[0] * (1.f / DM) + 1e-5f);
  float o0 = d0 * rstd * w[tid]       + b[tid];
  float o1 = d1 * rstd * w[tid + 256] + b[tid + 256];
  size_t fwd = (size_t)token * DM;
  size_t bwd = (size_t)(bb * SEQ_L + (SEQ_L - 1 - l)) * DM;
  xn[fwd + tid]         = (_Float16)o0;
  xn[fwd + tid + 256]   = (_Float16)o1;
  xnrev[bwd + tid]       = (_Float16)o0;
  xnrev[bwd + tid + 256] = (_Float16)o1;
}

// Final: out = LN(x + y_f + flip(y_b)) with ln2 params, f32 out.
__global__ __launch_bounds__(256) void ln_out_kernel(
    const float* __restrict__ x, const float* __restrict__ yf, const float* __restrict__ yb,
    const float* __restrict__ w, const float* __restrict__ b, float* __restrict__ out)
{
  const int token = blockIdx.x;
  const int bb = token / SEQ_L, l = token % SEQ_L;
  const int tid = threadIdx.x;
  size_t fwd = (size_t)token * DM;
  size_t bwd = (size_t)(bb * SEQ_L + (SEQ_L - 1 - l)) * DM;
  float v0 = x[fwd + tid]       + yf[fwd + tid]       + yb[bwd + tid];
  float v1 = x[fwd + tid + 256] + yf[fwd + tid + 256] + yb[bwd + tid + 256];
  __shared__ float red[256];
  red[tid] = v0 + v1; __syncthreads();
  for (int s = 128; s > 0; s >>= 1) { if (tid < s) red[tid] += red[tid + s]; __syncthreads(); }
  const float mean = red[0] * (1.f / DM);
  __syncthreads();
  float d0 = v0 - mean, d1 = v1 - mean;
  red[tid] = d0 * d0 + d1 * d1; __syncthreads();
  for (int s = 128; s > 0; s >>= 1) { if (tid < s) red[tid] += red[tid + s]; __syncthreads(); }
  const float rstd = rsqrtf(red[0] * (1.f / DM) + 1e-5f);
  out[fwd + tid]       = d0 * rstd * w[tid]       + b[tid];
  out[fwd + tid + 256] = d1 * rstd * w[tid + 256] + b[tid + 256];
}

// ---------------------------------------------------------------------------
// Depthwise causal conv (k=4) + SiLU over the xc half of xz (f16 in/out).
// ---------------------------------------------------------------------------
__global__ __launch_bounds__(256) void conv_silu_kernel(
    const _Float16* __restrict__ xz, const float* __restrict__ cw,
    const float* __restrict__ cb, _Float16* __restrict__ xc)
{
  size_t id = (size_t)blockIdx.x * 256 + threadIdx.x;   // < MTOT*DI
  int c = (int)(id & (DI - 1));
  size_t tok = id >> 10;
  int bb = (int)(tok / SEQ_L), l = (int)(tok % SEQ_L);
  float acc = cb[c];
  #pragma unroll
  for (int k = 0; k < 4; ++k) {
    int ls = l + k - 3;
    if (ls >= 0)
      acc += cw[c * 4 + k] * (float)xz[((size_t)(bb * SEQ_L + ls)) * 2048 + c];
  }
  float s = acc / (1.f + __expf(-acc));   // SiLU
  xc[id] = (_Float16)s;
}

// ---------------------------------------------------------------------------
// dt = softplus(dt_r @ dtw^T + bias); K=32 so plain VALU. dtw pre-transposed
// to [32][1024] for coalesced reads. One token per block.
// ---------------------------------------------------------------------------
__global__ __launch_bounds__(256) void dt_kernel(
    const float* __restrict__ xdbl, const float* __restrict__ wt,
    const float* __restrict__ bias, float* __restrict__ dt)
{
  const int token = blockIdx.x;
  const int tid = threadIdx.x;
  __shared__ float s[RK];
  if (tid < RK) s[tid] = xdbl[(size_t)token * 64 + tid];
  __syncthreads();
  for (int j = tid; j < DI; j += 256) {
    float acc = bias[j];
    #pragma unroll
    for (int r2 = 0; r2 < RK; ++r2) acc += s[r2] * wt[r2 * DI + j];
    dt[(size_t)token * DI + j] = acc > 20.f ? acc : log1pf(__expf(acc));
  }
}

// ---------------------------------------------------------------------------
// Chunked selective scan (exact for the diagonal SSM: the chunk transition is
// exp(A * sum(dt))).  Phase 1: local scans from h=0 -> per-chunk (q, sum_dt).
// Phase 2: 32-step sequential combine per channel -> each chunk's true h_in
// (stored in-place over q).  Phase 3: local re-scan from h_in, fused gating.
// qbuf layout: [b][chunk][n][c]  (c fastest => coalesced).
// ---------------------------------------------------------------------------
__global__ __launch_bounds__(256) void scan_p1_kernel(
    const float* __restrict__ xdbl, const float* __restrict__ dt,
    const _Float16* __restrict__ xc, const float* __restrict__ A_log,
    float* __restrict__ qbuf, float* __restrict__ sdtbuf)
{
  const int k  = blockIdx.x;                  // chunk
  const int bb = blockIdx.y >> 2;
  const int c  = ((blockIdx.y & 3) << 8) + threadIdx.x;
  float A[DSTATE], h[DSTATE];
  #pragma unroll
  for (int n = 0; n < DSTATE; ++n) { A[n] = -__expf(A_log[c * DSTATE + n]); h[n] = 0.f; }
  float sdt = 0.f;
  for (int t = k * CLEN; t < (k + 1) * CLEN; ++t) {
    const size_t tok = (size_t)bb * SEQ_L + t;
    const float* bc = xdbl + tok * 64 + RK;   // uniform across the wave
    const size_t idx = tok * DI + c;
    const float dtv = dt[idx];
    const float u   = (float)xc[idx];
    sdt += dtv;
    #pragma unroll
    for (int n = 0; n < DSTATE; ++n)
      h[n] = __expf(dtv * A[n]) * h[n] + dtv * bc[n] * u;
  }
  const size_t base = ((size_t)(bb * NCHUNK + k) * DSTATE) * DI + c;
  #pragma unroll
  for (int n = 0; n < DSTATE; ++n) qbuf[base + (size_t)n * DI] = h[n];
  sdtbuf[(size_t)(bb * NCHUNK + k) * DI + c] = sdt;
}

__global__ __launch_bounds__(256) void scan_p2_kernel(
    float* __restrict__ qbuf, const float* __restrict__ sdtbuf,
    const float* __restrict__ A_log)
{
  const int bb = blockIdx.x >> 2;
  const int c  = ((blockIdx.x & 3) << 8) + threadIdx.x;
  float A[DSTATE], carry[DSTATE];
  #pragma unroll
  for (int n = 0; n < DSTATE; ++n) { A[n] = -__expf(A_log[c * DSTATE + n]); carry[n] = 0.f; }
  for (int k = 0; k < NCHUNK; ++k) {
    const size_t base = ((size_t)(bb * NCHUNK + k) * DSTATE) * DI + c;
    const float sdt = sdtbuf[(size_t)(bb * NCHUNK + k) * DI + c];
    #pragma unroll
    for (int n = 0; n < DSTATE; ++n) {
      const float qn = qbuf[base + (size_t)n * DI];
      qbuf[base + (size_t)n * DI] = carry[n];            // h_in of chunk k
      carry[n] = __expf(A[n] * sdt) * carry[n] + qn;     // h at chunk end
    }
  }
}

__global__ __launch_bounds__(256) void scan_p3_kernel(
    const float* __restrict__ xdbl, const float* __restrict__ dt,
    const _Float16* __restrict__ xc, const _Float16* __restrict__ xz,
    const float* __restrict__ A_log, const float* __restrict__ Dp,
    const float* __restrict__ qbuf /* h_in */, _Float16* __restrict__ yg)
{
  const int k  = blockIdx.x;
  const int bb = blockIdx.y >> 2;
  const int c  = ((blockIdx.y & 3) << 8) + threadIdx.x;
  float A[DSTATE], h[DSTATE];
  const size_t base = ((size_t)(bb * NCHUNK + k) * DSTATE) * DI + c;
  #pragma unroll
  for (int n = 0; n < DSTATE; ++n) {
    A[n] = -__expf(A_log[c * DSTATE + n]);
    h[n] = qbuf[base + (size_t)n * DI];
  }
  const float Dc = Dp[c];
  for (int t = k * CLEN; t < (k + 1) * CLEN; ++t) {
    const size_t tok = (size_t)bb * SEQ_L + t;
    const float* bc = xdbl + tok * 64 + RK;
    const size_t idx = tok * DI + c;
    const float dtv = dt[idx];
    const float u   = (float)xc[idx];
    float y = 0.f;
    #pragma unroll
    for (int n = 0; n < DSTATE; ++n) {
      h[n] = __expf(dtv * A[n]) * h[n] + dtv * bc[n] * u;
      y += h[n] * bc[DSTATE + n];
    }
    y += u * Dc;
    const float z = (float)xz[tok * 2048 + DI + c];
    const float g = z / (1.f + __expf(-z));
    yg[idx] = (_Float16)(y * g);
  }
}

// ---------------------------------------------------------------------------
// Weight prep
// ---------------------------------------------------------------------------
__global__ __launch_bounds__(256) void f32_to_f16_kernel(
    const float* __restrict__ src, _Float16* __restrict__ dst, int n)
{
  int i = blockIdx.x * 256 + threadIdx.x;
  if (i < n) dst[i] = (_Float16)src[i];
}

__global__ __launch_bounds__(256) void dtw_transpose_kernel(
    const float* __restrict__ w /*(2,1024,32)*/, float* __restrict__ wt /*(2,32,1024)*/)
{
  int i = blockIdx.x * 256 + threadIdx.x;           // < 2*1024*32
  if (i < 2 * DI * RK) {
    int d = i / (DI * RK), rem = i % (DI * RK);
    int row = rem / RK, r = rem % RK;
    wt[(size_t)d * RK * DI + r * DI + row] = w[i];
  }
}

// ---------------------------------------------------------------------------
extern "C" void kernel_launch(void* const* d_in, const int* in_sizes, int n_in,
                              void* d_out, int out_size, void* d_ws, size_t ws_size,
                              hipStream_t stream) {
  const float* x      = (const float*)d_in[0];
  const float* ln1_w  = (const float*)d_in[1];
  const float* ln1_b  = (const float*)d_in[2];
  const float* ln2_w  = (const float*)d_in[3];
  const float* ln2_b  = (const float*)d_in[4];
  const float* in_w   = (const float*)d_in[5];   // (2, 2048, 512)
  const float* conv_w = (const float*)d_in[6];   // (2, 1024, 4)
  const float* conv_b = (const float*)d_in[7];   // (2, 1024)
  const float* xp_w   = (const float*)d_in[8];   // (2, 64, 1024)
  const float* dtp_w  = (const float*)d_in[9];   // (2, 1024, 32)
  const float* dtp_b  = (const float*)d_in[10];  // (2, 1024)
  const float* A_log  = (const float*)d_in[11];  // (2, 1024, 16)
  const float* Dp     = (const float*)d_in[12];  // (2, 1024)
  const float* out_w  = (const float*)d_in[13];  // (2, 512, 1024)
  float* out = (float*)d_out;

  char* p = (char*)d_ws;
  auto alloc = [&](size_t bytes) -> void* {
    void* r = (void*)p;
    p += (bytes + 255) & ~(size_t)255;
    return r;
  };
  _Float16* xn16    = (_Float16*)alloc((size_t)MTOT * DM * 2);
  _Float16* xnr16   = (_Float16*)alloc((size_t)MTOT * DM * 2);
  _Float16* w_in16  = (_Float16*)alloc((size_t)2 * 2048 * 512 * 2);
  _Float16* w_xp16  = (_Float16*)alloc((size_t)2 * 64 * 1024 * 2);
  _Float16* w_out16 = (_Float16*)alloc((size_t)2 * 512 * 1024 * 2);
  float*    dtwT    = (float*)   alloc((size_t)2 * RK * DI * 4);
  _Float16* xz16    = (_Float16*)alloc((size_t)MTOT * 2048 * 2);
  _Float16* xc16    = (_Float16*)alloc((size_t)MTOT * DI * 2);
  float*    xdbl    = (float*)   alloc((size_t)MTOT * 64 * 4);
  float*    dtb     = (float*)   alloc((size_t)MTOT * DI * 4);
  _Float16* yg16    = (_Float16*)alloc((size_t)MTOT * DI * 2);
  float*    yfb     = (float*)   alloc((size_t)MTOT * DM * 4);
  float*    ybb     = (float*)   alloc((size_t)MTOT * DM * 4);
  float*    qbuf    = (float*)   alloc((size_t)B_SZ * NCHUNK * DSTATE * DI * 4);
  float*    sdtbuf  = (float*)   alloc((size_t)B_SZ * NCHUNK * DI * 4);
  (void)ws_size; (void)in_sizes; (void)n_in; (void)out_size;

  // --- weight prep (f32 -> f16, dt_proj transpose) ---
  {
    int n1 = 2 * 2048 * 512, n2 = 2 * 64 * 1024, n3 = 2 * 512 * 1024, n4 = 2 * DI * RK;
    f32_to_f16_kernel<<<(n1 + 255) / 256, 256, 0, stream>>>(in_w,  w_in16,  n1);
    f32_to_f16_kernel<<<(n2 + 255) / 256, 256, 0, stream>>>(xp_w,  w_xp16,  n2);
    f32_to_f16_kernel<<<(n3 + 255) / 256, 256, 0, stream>>>(out_w, w_out16, n3);
    dtw_transpose_kernel<<<(n4 + 255) / 256, 256, 0, stream>>>(dtp_w, dtwT);
  }

  // --- LN1 -> f16 activations (forward + flipped) ---
  ln_in_kernel<<<MTOT, 256, 0, stream>>>(x, ln1_w, ln1_b, xn16, xnr16);

  for (int dir = 0; dir < 2; ++dir) {
    const _Float16* Act  = dir ? xnr16 : xn16;
    const float*    Adir = A_log + (size_t)dir * DI * DSTATE;
    // in_proj: 8192 x 2048 x 512 -> xz (f16)
    gemm_wmma<128, true><<<dim3(MTOT / 128, 2048 / 128), 256, 0, stream>>>(
        Act, w_in16 + (size_t)dir * 2048 * 512, xz16, 512, 512, 512, 2048);
    // conv + silu
    conv_silu_kernel<<<(MTOT * DI) / 256, 256, 0, stream>>>(
        xz16, conv_w + (size_t)dir * DI * 4, conv_b + (size_t)dir * DI, xc16);
    // x_proj: 8192 x 64 x 1024 -> x_dbl (f32)
    gemm_wmma<64, false><<<dim3(MTOT / 128, 1), 256, 0, stream>>>(
        xc16, w_xp16 + (size_t)dir * 64 * 1024, xdbl, 1024, 1024, 1024, 64);
    // dt
    dt_kernel<<<MTOT, 256, 0, stream>>>(
        xdbl, dtwT + (size_t)dir * RK * DI, dtp_b + (size_t)dir * DI, dtb);
    // chunked selective scan + gating -> yg (f16)
    scan_p1_kernel<<<dim3(NCHUNK, B_SZ * 4), 256, 0, stream>>>(
        xdbl, dtb, xc16, Adir, qbuf, sdtbuf);
    scan_p2_kernel<<<B_SZ * 4, 256, 0, stream>>>(qbuf, sdtbuf, Adir);
    scan_p3_kernel<<<dim3(NCHUNK, B_SZ * 4), 256, 0, stream>>>(
        xdbl, dtb, xc16, xz16, Adir, Dp + (size_t)dir * DI, qbuf, yg16);
    // out_proj: 8192 x 512 x 1024 -> y_dir (f32)
    gemm_wmma<128, false><<<dim3(MTOT / 128, 512 / 128), 256, 0, stream>>>(
        yg16, w_out16 + (size_t)dir * 512 * 1024, dir ? ybb : yfb,
        1024, 1024, 1024, 512);
  }

  // --- residual + LN2 (un-flips backward contribution) ---
  ln_out_kernel<<<MTOT, 256, 0, stream>>>(x, yfb, ybb, ln2_w, ln2_b, out);
}
